// DirichletQuantizer_60284160967206
// MI455X (gfx1250) — matrix operand order repaired and checked by reading, not compiled, not run on hardware
//
#include <hip/hip_runtime.h>
#include <math.h>

typedef __attribute__((ext_vector_type(16))) _Float16 v16h;
typedef __attribute__((ext_vector_type(8)))  float    v8f;

#define Bdim 16
#define Ndim 1024
#define Ddim 256
#define HWdim 1024
#define PIX  16384            // B*H*W
#define TEMP_INV 2.0f         // 1/0.5
#define KL_W 1.0e-3f
#define NREP 32               // accumulator replicas (atomic contention relief)

// workspace layout (bytes)
//   [0, 33554432)           A-fragment f16 staging: PIX*Ndim halfs (WMMA A layout)
//   [33554432, 34078720)    codebook^T f16: Ddim*Ndim halfs
//   [34078720, +131200)     accumulators: avg_probs[NREP][1024] f32, kl_sum[NREP] f32
#define WS_CBT_OFF  33554432ull
#define WS_ACC_OFF  34078720ull
#define ACC_COUNT   (NREP * Ndim + NREP)

// ---------------------------------------------------------------- digamma
__device__ __forceinline__ float digammaf_dev(float x) {
    float r = 0.f;
    while (x < 6.f) { r -= 1.f / x; x += 1.f; }
    float inv = 1.f / x;
    float f = inv * inv;
    return r + __logf(x) - 0.5f * inv
             - f * (1.f/12.f - f * (1.f/120.f - f * (1.f/252.f)));
}

// ---------------------------------------------------------------- zero accumulators
__global__ void dq_zero_acc(float* acc) {
    int t = blockIdx.x * blockDim.x + threadIdx.x;
    if (t < ACC_COUNT) acc[t] = 0.f;
}

// ---------------------------------------------------------------- codebook [N,D] f32 -> [D,N] f16
__global__ void dq_cvt_codebook(const float* __restrict__ cb, _Float16* __restrict__ cbT) {
    int tid = blockIdx.x * blockDim.x + threadIdx.x;   // Ddim*Ndim threads
    int d = tid >> 10;
    int n = tid & (Ndim - 1);
    cbT[tid] = (_Float16)cb[n * Ddim + d];
}

// ---------------------------------------------------------------- per-pixel pass
// 1 block = 1 pixel, 256 threads x 4 channels
__global__ void dq_pixel_kernel(const float* __restrict__ logits,
                                const float* __restrict__ noise,
                                _Float16* __restrict__ Aws,
                                float* __restrict__ acc,
                                float* __restrict__ out_idx)
{
    __shared__ float red[256];
    __shared__ int   redi[256];
    __shared__ float sbroad[4];                 // S, digamma(S), ymax, denom
    __shared__ __align__(16) float smp[Ndim];   // sample values for this pixel

    const int t  = threadIdx.x;
    const int p  = blockIdx.x;                  // pixel id = b*HW + hw
    const int b  = p >> 10;
    const int hw = p & (HWdim - 1);
    const int rep = p & (NREP - 1);

    const float* lg = logits + (size_t)b * Ndim * HWdim + hw;   // stride HW between n
    const float* un = noise  + (size_t)p * Ndim;                // contiguous in n

    // alpha = softplus(logit) + 1
    float alpha[4];
#pragma unroll
    for (int j = 0; j < 4; ++j) {
        int n = t + j * 256;
        float x = lg[(size_t)n * HWdim];
        float sp = (x > 20.f) ? x : log1pf(__expf(x));
        alpha[j] = sp + 1.f;
    }

    // reduce S = sum(alpha)
    red[t] = alpha[0] + alpha[1] + alpha[2] + alpha[3];
    __syncthreads();
    for (int s = 128; s > 0; s >>= 1) {
        if (t < s) red[t] += red[t + s];
        __syncthreads();
    }
    if (t == 0) { sbroad[0] = red[0]; sbroad[1] = digammaf_dev(red[0]); }
    __syncthreads();
    const float S    = sbroad[0];
    const float dgS  = sbroad[1];
    const float logS = __logf(S);

    // y = (log(alpha/S) + gumbel)/TEMP ; track max+arg
    float y[4]; float mymax = -1e30f; int myarg = 0;
#pragma unroll
    for (int j = 0; j < 4; ++j) {
        int n = t + j * 256;
        float g = -__logf(-__logf(un[n]));
        float yy = (__logf(alpha[j]) - logS + g) * TEMP_INV;
        y[j] = yy;
        if (yy > mymax) { mymax = yy; myarg = n; }
    }
    red[t] = mymax; redi[t] = myarg;
    __syncthreads();
    for (int s = 128; s > 0; s >>= 1) {
        if (t < s && red[t + s] > red[t]) { red[t] = red[t + s]; redi[t] = redi[t + s]; }
        __syncthreads();
    }
    if (t == 0) sbroad[2] = red[0];
    __syncthreads();
    const float ymax = sbroad[2];
    const int   argn = redi[0];

    // softmax denom
    float e[4]; float esum = 0.f;
#pragma unroll
    for (int j = 0; j < 4; ++j) { e[j] = __expf(y[j] - ymax); esum += e[j]; }
    __syncthreads();
    red[t] = esum;
    __syncthreads();
    for (int s = 128; s > 0; s >>= 1) {
        if (t < s) red[t] += red[t + s];
        __syncthreads();
    }
    if (t == 0) sbroad[3] = red[0];
    __syncthreads();
    const float inv = 1.f / sbroad[3];

    // stage sample in LDS + replicated avg_probs accumulation
#pragma unroll
    for (int j = 0; j < 4; ++j) {
        int n = t + j * 256;
        float s = e[j] * inv;
        smp[n] = s;
        atomicAdd(&acc[rep * Ndim + n], s);
    }
    __syncthreads();

    // vectorized A-fragment stores: for fixed (kb, lane-half hh) the 16 halfs
    // at Aws[((tm*32+kb)*32 + hh*16 + m)*16 + 0..15] come from the contiguous
    // 16-float run smp[kb*32+hh*16 .. +15].
    const int tm = p >> 4;
    const int m  = p & 15;
    if (t < 64) {
        int kb = t >> 1;
        int hh = t & 1;
        int base0 = kb * 32 + hh * 16;
        float4 x0 = *(const float4*)&smp[base0];
        float4 x1 = *(const float4*)&smp[base0 + 4];
        float4 x2 = *(const float4*)&smp[base0 + 8];
        float4 x3 = *(const float4*)&smp[base0 + 12];
        v16h f;
        f[0]=(_Float16)x0.x; f[1]=(_Float16)x0.y; f[2]=(_Float16)x0.z; f[3]=(_Float16)x0.w;
        f[4]=(_Float16)x1.x; f[5]=(_Float16)x1.y; f[6]=(_Float16)x1.z; f[7]=(_Float16)x1.w;
        f[8]=(_Float16)x2.x; f[9]=(_Float16)x2.y; f[10]=(_Float16)x2.z; f[11]=(_Float16)x2.w;
        f[12]=(_Float16)x3.x; f[13]=(_Float16)x3.y; f[14]=(_Float16)x3.z; f[15]=(_Float16)x3.w;
        ((v16h*)Aws)[((size_t)tm * 32 + kb) * 32 + hh * 16 + m] = f;
    }

    // KL partials
    float partial = 0.f;
#pragma unroll
    for (int j = 0; j < 4; ++j) {
        partial += (alpha[j] - 1.f) * (digammaf_dev(alpha[j]) - dgS);
        partial -= lgammaf(alpha[j]);
    }
    __syncthreads();
    red[t] = partial;
    __syncthreads();
    for (int s = 128; s > 0; s >>= 1) {
        if (t < s) red[t] += red[t + s];
        __syncthreads();
    }
    if (t == 0) {
        float kl = lgammaf(S) - lgammaf((float)Ndim) + red[0];
        atomicAdd(&acc[NREP * Ndim + rep], kl);
        out_idx[p] = (float)argn;
    }
}

// ---------------------------------------------------------------- WMMA GEMM
// z_q[bhw, d] = sum_n sample[bhw,n]*codebook[n,d] ; out laid out as [B,D,H,W].
// 1 wave computes a 16x64 tile: one A fragment reused by 4 B fragments.
// Software-pipelined: iteration k+1 fragment loads are issued before the
// iteration-k wmmas so waits on loadcnt are partial, not full drains.
__global__ void __launch_bounds__(128) dq_gemm_wmma(const _Float16* __restrict__ Aws,
                                                    const _Float16* __restrict__ cbT,
                                                    float* __restrict__ zq)
{
    const int wave = threadIdx.x >> 5;
    const int lane = threadIdx.x & 31;
    const int tile = blockIdx.x * 4 + wave;   // 4096 tiles total
    const int tg = tile & 3;                  // 64-column group (D/64)
    const int tm = tile >> 2;                 // 1024 row tiles (PIX/16)
    const int m    = lane & 15;
    const int half = lane >> 4;

    // A fragment: v16h index = (tm*32 + kb)*32 + lane (contiguous 32B per lane)
    const v16h* aptr = (const v16h*)Aws + ((size_t)tm * 32) * 32 + lane;
    // B fragments for column tiles tg*4 + i
    const v16h* b0 = (const v16h*)cbT + ((size_t)((tg * 4 + 0) * 16 + m) * 64 + half);
    const v16h* b1 = (const v16h*)cbT + ((size_t)((tg * 4 + 1) * 16 + m) * 64 + half);
    const v16h* b2 = (const v16h*)cbT + ((size_t)((tg * 4 + 2) * 16 + m) * 64 + half);
    const v16h* b3 = (const v16h*)cbT + ((size_t)((tg * 4 + 3) * 16 + m) * 64 + half);

    v8f c0 = {}, c1 = {}, c2 = {}, c3 = {};

    // prologue: load kb=0 fragments
    v16h a  = aptr[0];
    v16h f0 = b0[0];
    v16h f1 = b1[0];
    v16h f2 = b2[0];
    v16h f3 = b3[0];

#pragma unroll 2
    for (int kb = 0; kb < 32; ++kb) {
        const size_t nk = (size_t)((kb + 1) & 31);   // wrap: last prefetch is redundant, in-bounds
        v16h an = aptr[nk * 32];
        v16h g0 = b0[nk * 2];
        v16h g1 = b1[nk * 2];
        v16h g2 = b2[nk * 2];
        v16h g3 = b3[nk * 2];
        c0 = __builtin_amdgcn_wmma_f32_16x16x32_f16(false, a, false, f0, (short)0, c0, false, false);
        c1 = __builtin_amdgcn_wmma_f32_16x16x32_f16(false, a, false, f1, (short)0, c1, false, false);
        c2 = __builtin_amdgcn_wmma_f32_16x16x32_f16(false, a, false, f2, (short)0, c2, false, false);
        c3 = __builtin_amdgcn_wmma_f32_16x16x32_f16(false, a, false, f3, (short)0, c3, false, false);
        a = an; f0 = g0; f1 = g1; f2 = g2; f3 = g3;
    }

    // C/D layout: lane column = lane&15, VGPR r -> row m = r + 8*half.
    // rows map to contiguous hw -> two contiguous float4 stores per lane per tile.
    const int grow0 = tm * 16 + half * 8;
    const int bb    = grow0 >> 10;              // tile never straddles batches
    const int hw0   = grow0 & (HWdim - 1);
    {
        float* o = zq + (((size_t)bb * Ddim + ((tg * 4 + 0) * 16 + m)) * HWdim + hw0);
        *(float4*)(o)     = make_float4(c0[0], c0[1], c0[2], c0[3]);
        *(float4*)(o + 4) = make_float4(c0[4], c0[5], c0[6], c0[7]);
    }
    {
        float* o = zq + (((size_t)bb * Ddim + ((tg * 4 + 1) * 16 + m)) * HWdim + hw0);
        *(float4*)(o)     = make_float4(c1[0], c1[1], c1[2], c1[3]);
        *(float4*)(o + 4) = make_float4(c1[4], c1[5], c1[6], c1[7]);
    }
    {
        float* o = zq + (((size_t)bb * Ddim + ((tg * 4 + 2) * 16 + m)) * HWdim + hw0);
        *(float4*)(o)     = make_float4(c2[0], c2[1], c2[2], c2[3]);
        *(float4*)(o + 4) = make_float4(c2[4], c2[5], c2[6], c2[7]);
    }
    {
        float* o = zq + (((size_t)bb * Ddim + ((tg * 4 + 3) * 16 + m)) * HWdim + hw0);
        *(float4*)(o)     = make_float4(c3[0], c3[1], c3[2], c3[3]);
        *(float4*)(o + 4) = make_float4(c3[4], c3[5], c3[6], c3[7]);
    }
}

// ---------------------------------------------------------------- finalize
__global__ void dq_finalize(const float* __restrict__ acc,
                            float* __restrict__ out_kl,
                            float* __restrict__ out_perp)
{
    __shared__ float red[1024];
    const int t = threadIdx.x;
    float sum = 0.f;
#pragma unroll
    for (int r = 0; r < NREP; ++r) sum += acc[r * Ndim + t];
    float ap = sum * (1.f / (float)PIX);
    red[t] = ap * __logf(ap + 1e-10f);
    __syncthreads();
    for (int s = 512; s > 0; s >>= 1) {
        if (t < s) red[t] += red[t + s];
        __syncthreads();
    }
    if (t == 0) {
        float kl = 0.f;
#pragma unroll
        for (int r = 0; r < NREP; ++r) kl += acc[NREP * Ndim + r];
        *out_perp = __expf(-red[0]);
        *out_kl   = KL_W * kl * (1.f / (float)PIX);
    }
}

// ---------------------------------------------------------------- launch
extern "C" void kernel_launch(void* const* d_in, const int* in_sizes, int n_in,
                              void* d_out, int out_size, void* d_ws, size_t ws_size,
                              hipStream_t stream) {
    const float* logits   = (const float*)d_in[0];   // [B,N,H,W]
    const float* codebook = (const float*)d_in[1];   // [N,D]
    const float* noise    = (const float*)d_in[2];   // [B,H,W,N]

    float* out = (float*)d_out;
    float* zq      = out;                          // 4194304 floats
    float* out_kl  = out + 4194304;                // 1
    float* out_idx = out + 4194305;                // 16384
    float* out_pp  = out + 4210689;                // 1

    _Float16* Aws = (_Float16*)d_ws;
    _Float16* cbT = (_Float16*)((char*)d_ws + WS_CBT_OFF);
    float*    acc = (float*)((char*)d_ws + WS_ACC_OFF);

    dq_zero_acc<<<(ACC_COUNT + 255) / 256, 256, 0, stream>>>(acc);
    dq_cvt_codebook<<<(Ddim * Ndim) / 256, 256, 0, stream>>>(codebook, cbT);
    dq_pixel_kernel<<<PIX, 256, 0, stream>>>(logits, noise, Aws, acc, out_idx);
    dq_gemm_wmma<<<1024, 128, 0, stream>>>(Aws, cbT, zq);
    dq_finalize<<<1, 1024, 0, stream>>>(acc, out_kl, out_pp);
}
